// ChunkAttention_48567490183176
// MI455X (gfx1250) — matrix operand order repaired and checked by reading, not compiled, not run on hardware
//
#include <hip/hip_runtime.h>

#define CHUNK  64
#define NFEAT  256
#define NHEAD  8
#define DK     32
#define SCALE  0.17677669529663687f   // 1/sqrt(32)

#define XLD  264   // staging / q / k row stride (bf16 elems), mult of 8, bank-spread
#define VTLD 72    // v-transposed row stride
#define SLD  65    // f32 score row stride
#define ALD  72    // attn-prob (bf16) row stride, overlays sbuf

typedef __attribute__((ext_vector_type(16))) __bf16       v16bf;
typedef __attribute__((ext_vector_type(8)))  float        v8f;
typedef __attribute__((ext_vector_type(4)))  unsigned int u32x4;

struct Frag { union { v16bf v; u32x4 u[2]; }; };
struct Acc  { union { v8f  v; float f[8]; }; };

__device__ inline unsigned short f2bf(float x) {
  unsigned int u = __float_as_uint(x);
  u += 0x7FFFu + ((u >> 16) & 1u);          // round-to-nearest-even
  return (unsigned short)(u >> 16);
}

// A fragment: 16x32 bf16 tile from row-major (M x K) storage.
// lanes 0-15: K 0..7 & 16..23 ; lanes 16-31: K 8..15 & 24..31 (ISA 16-bit A layout)
__device__ inline Frag frag_a(const unsigned short* p, int ld, int r0, int c0, int lane) {
  Frag f;
  const unsigned short* q = p + (r0 + (lane & 15)) * ld + c0 + ((lane >> 4) << 3);
  f.u[0] = *(const u32x4*)q;
  f.u[1] = *(const u32x4*)(q + 16);
  return f;
}
// B fragment: 32x16 bf16 tile, loaded from transposed (N x K row-major) storage.
// lanes 0-15: K 0..7 & 16..23 ; lanes 16-31: K 8..15 & 24..31 (matches A)
__device__ inline Frag frag_b(const unsigned short* p, int ld, int n0, int c0, int lane) {
  Frag f;
  const unsigned short* q = p + (n0 + (lane & 15)) * ld + c0 + ((lane >> 4) << 3);
  f.u[0] = *(const u32x4*)q;
  f.u[1] = *(const u32x4*)(q + 16);
  return f;
}

__device__ inline v8f wmma_bf16(const Frag& a, const Frag& b, v8f c) {
  return __builtin_amdgcn_wmma_f32_16x16x32_bf16(false, a.v, false, b.v,
                                                 (short)0, c, false, false);
}

// ---------------------------------------------------------------------------
// Weight f32 -> bf16 pre-pass (4 x 256x256 into d_ws)
// ---------------------------------------------------------------------------
__global__ void ChunkAttention_wconv(const float* __restrict__ wq,
                                     const float* __restrict__ wk,
                                     const float* __restrict__ wv,
                                     const float* __restrict__ wo,
                                     unsigned short* __restrict__ dst) {
  int i = blockIdx.x * blockDim.x + threadIdx.x;
  if (i >= 4 * NFEAT * NFEAT) return;
  int which = i >> 16, off = i & 0xFFFF;
  const float* s = (which == 0) ? wq : (which == 1) ? wk : (which == 2) ? wv : wo;
  dst[i] = f2bf(s[off]);
}

// ---------------------------------------------------------------------------
// Fused per-chunk attention: 1 block = 1 chunk of 64 tokens, 256 threads
// ---------------------------------------------------------------------------
__global__ __launch_bounds__(256, 1)
void ChunkAttention_48567490183176_kernel(
    const float* __restrict__ gq, const float* __restrict__ gk,
    const float* __restrict__ gv,
    const float* __restrict__ mask,  const float* __restrict__ cmask,
    const float* __restrict__ bq,    const float* __restrict__ bk,
    const float* __restrict__ bv,    const float* __restrict__ bo,
    const float* __restrict__ rpb,
    const unsigned short* __restrict__ wq, const unsigned short* __restrict__ wk,
    const unsigned short* __restrict__ wv, const unsigned short* __restrict__ wo,
    float* __restrict__ out, int nck)
{
  __shared__ __align__(16) unsigned short xbuf[CHUNK * XLD];   // stage / attn-out
  __shared__ __align__(16) unsigned short qbuf[CHUNK * XLD];
  __shared__ __align__(16) unsigned short kbuf[CHUNK * XLD];
  __shared__ __align__(16) unsigned short vtbuf[NFEAT * VTLD]; // v^T per head
  __shared__ __align__(16) float          sbuf[CHUNK * SLD];   // scores (f32)
  unsigned short* abuf = (unsigned short*)sbuf;                // overlays sbuf

  const int tid  = threadIdx.x;
  const int wave = tid >> 5;
  const int lane = tid & 31;
  const int b    = blockIdx.x;
  const int moff = (lane >> 4) << 3;   // D-matrix row offset for upper half-wave

  auto stage = [&](const float* __restrict__ src) {
    for (int e = tid; e < CHUNK * NFEAT; e += 256) {
      int r = e >> 8, c = e & 255;
      xbuf[r * XLD + c] = f2bf(src[e]);
    }
  };

  // y = x @ W^T + bias ; W row-major (N x K) bf16, x staged in xbuf.
  auto proj = [&](const unsigned short* __restrict__ W, const float* __restrict__ bias,
                  unsigned short* __restrict__ dst, bool transpose_out) {
    for (int t = wave; t < 64; t += 8) {
      int r0 = (t >> 4) << 4, c0 = (t & 15) << 4;
      int n = c0 + (lane & 15);
      float bval = bias[n];
      Acc acc;
      #pragma unroll
      for (int i = 0; i < 8; ++i) acc.f[i] = bval;
      #pragma unroll
      for (int k0 = 0; k0 < NFEAT; k0 += 32) {
        Frag a  = frag_a(xbuf, XLD, r0, k0, lane);
        Frag bb = frag_b(W, NFEAT, c0, k0, lane);
        acc.v = wmma_bf16(a, bb, acc.v);
      }
      #pragma unroll
      for (int i = 0; i < 8; ++i) {
        int m = r0 + i + moff;
        if (transpose_out) dst[n * VTLD + m] = f2bf(acc.f[i]);  // v^T: [feat][token]
        else               dst[m * XLD + n]  = f2bf(acc.f[i]);
      }
    }
  };

  const size_t cb = (size_t)b * CHUNK * NFEAT;

  stage(gq + cb); __syncthreads();
  proj(wq, bq, qbuf, false); __syncthreads();
  stage(gk + cb); __syncthreads();
  proj(wk, bk, kbuf, false); __syncthreads();
  stage(gv + cb); __syncthreads();
  proj(wv, bv, vtbuf, true); __syncthreads();

  const float* mrow = mask  + (size_t)b * CHUNK;
  const float* crow = cmask + (size_t)(b % nck) * CHUNK * CHUNK;

  for (int h = 0; h < NHEAD; ++h) {
    // ---- scores = q_h @ k_h^T (16 tiles of 16x16, K = 32 -> one WMMA each)
    for (int t = wave; t < 16; t += 8) {
      int r0 = (t >> 2) << 4, c0 = (t & 3) << 4;
      Acc acc;
      #pragma unroll
      for (int i = 0; i < 8; ++i) acc.f[i] = 0.f;
      Frag a  = frag_a(qbuf, XLD, r0, h * DK, lane);
      Frag bb = frag_b(kbuf, XLD, c0, h * DK, lane);   // kbuf is N x K for B
      acc.v = wmma_bf16(a, bb, acc.v);
      #pragma unroll
      for (int i = 0; i < 8; ++i)
        sbuf[(r0 + i + moff) * SLD + c0 + (lane & 15)] = acc.f[i];
    }
    __syncthreads();

    // ---- masking + softmax, one row per thread (held fully in VGPRs)
    {
      const bool active = (tid < CHUNK);
      float r[CHUNK];
      float inv = 0.f;
      if (active) {
        const int i = tid;
        const float* sr = sbuf + i * SLD;
        float mx = -1e30f;
        #pragma unroll
        for (int j = 0; j < CHUNK; ++j) {
          float s = sr[j] * SCALE + rpb[(i - j + CHUNK - 1) * NHEAD + h];
          s *= mrow[j];                        // multiplicative padding mask
          s = (s == 0.0f) ? -100.0f : s;       // torch-source quirk
          s += crow[i * CHUNK + j];            // additive shifted-chunk mask
          r[j] = s;
          mx = fmaxf(mx, s);
        }
        float sum = 0.f;
        #pragma unroll
        for (int j = 0; j < CHUNK; ++j) { float e = __expf(r[j] - mx); r[j] = e; sum += e; }
        inv = 1.0f / sum;
      }
      __syncthreads();                         // sbuf reads drained before overlay write
      if (active) {
        #pragma unroll
        for (int j = 0; j < CHUNK; ++j) abuf[tid * ALD + j] = f2bf(r[j] * inv);
      }
    }
    __syncthreads();

    // ---- attn @ v_h (8 tiles, one per wave; K = 64 -> two WMMAs)
    {
      int r0 = (wave >> 1) << 4, c0 = (wave & 1) << 4;
      Acc acc;
      #pragma unroll
      for (int i = 0; i < 8; ++i) acc.f[i] = 0.f;
      #pragma unroll
      for (int k0 = 0; k0 < CHUNK; k0 += 32) {
        Frag a  = frag_a(abuf, ALD, r0, k0, lane);
        Frag bb = frag_b(vtbuf, VTLD, h * DK + c0, k0, lane);
        acc.v = wmma_bf16(a, bb, acc.v);
      }
      int n = h * DK + c0 + (lane & 15);
      #pragma unroll
      for (int i = 0; i < 8; ++i)
        xbuf[(r0 + i + moff) * XLD + n] = f2bf(acc.f[i]);   // concat heads
    }
    __syncthreads();
  }

  // ---- out = concat(heads) @ Wo^T + bo  -> global f32
  float* orow = out + cb;
  for (int t = wave; t < 64; t += 8) {
    int r0 = (t >> 4) << 4, c0 = (t & 15) << 4;
    int n = c0 + (lane & 15);
    float bval = bo[n];
    Acc acc;
    #pragma unroll
    for (int i = 0; i < 8; ++i) acc.f[i] = bval;
    #pragma unroll
    for (int k0 = 0; k0 < NFEAT; k0 += 32) {
      Frag a  = frag_a(xbuf, XLD, r0, k0, lane);
      Frag bb = frag_b(wo, NFEAT, c0, k0, lane);
      acc.v = wmma_bf16(a, bb, acc.v);
    }
    #pragma unroll
    for (int i = 0; i < 8; ++i)
      orow[(size_t)(r0 + i + moff) * NFEAT + n] = acc.f[i];
  }
}

extern "C" void kernel_launch(void* const* d_in, const int* in_sizes, int n_in,
                              void* d_out, int out_size, void* d_ws, size_t ws_size,
                              hipStream_t stream) {
  const float* gq    = (const float*)d_in[0];
  const float* gk    = (const float*)d_in[1];
  const float* gv    = (const float*)d_in[2];
  const float* mask  = (const float*)d_in[3];
  const float* cmask = (const float*)d_in[4];
  const float* Wq    = (const float*)d_in[5];
  const float* bq    = (const float*)d_in[6];
  const float* Wk    = (const float*)d_in[7];
  const float* bk    = (const float*)d_in[8];
  const float* Wv    = (const float*)d_in[9];
  const float* bv    = (const float*)d_in[10];
  const float* Wo    = (const float*)d_in[11];
  const float* bo    = (const float*)d_in[12];
  const float* rpb   = (const float*)d_in[13];
  float* out = (float*)d_out;

  unsigned short* wbf = (unsigned short*)d_ws;          // 4 x 65536 bf16 = 512 KB
  const int B   = in_sizes[0] / (CHUNK * NFEAT);        // 2048
  const int nck = in_sizes[4] / (CHUNK * CHUNK);        // 128

  ChunkAttention_wconv<<<(4 * NFEAT * NFEAT + 255) / 256, 256, 0, stream>>>(
      Wq, Wk, Wv, Wo, wbf);

  ChunkAttention_48567490183176_kernel<<<B, 256, 0, stream>>>(
      gq, gk, gv, mask, cmask, bq, bk, bv, bo, rpb,
      wbf, wbf + 65536, wbf + 2 * 65536, wbf + 3 * 65536, out, nck);
}
